// GATConvTransformerEncoderLayer_44641890075096
// MI455X (gfx1250) — matrix operand order repaired and checked by reading, not compiled, not run on hardware
//
#include <hip/hip_runtime.h>
#include <hip/hip_bf16.h>
#include <math.h>

// ---- problem constants (match reference) ----
#define NNODES 50000
#define NEDGES 800000
#define FD     128      // H*D channels (also K of every GEMM)
#define NHEAD  4
#define DHEAD  32
#define DFFOUT 512      // H*DFF
#define SLOPE_ATT 0.2f
#define SLOPE_ACT 0.01f

#define LDS_STRIDE 144  // halves per staged W column (128 + 16 pad, keeps 32B align)

typedef __attribute__((ext_vector_type(16))) _Float16 v16h;
typedef __attribute__((ext_vector_type(8)))  float    v8f;

// order-preserving float<->uint mapping for atomic max
__device__ __forceinline__ unsigned f2ord(float f) {
  unsigned u = __float_as_uint(f);
  return (u & 0x80000000u) ? ~u : (u | 0x80000000u);
}
__device__ __forceinline__ float ord2f(unsigned u) {
  return (u & 0x80000000u) ? __uint_as_float(u & 0x7fffffffu)
                           : __uint_as_float(~u);
}

// ---------------------------------------------------------------------------
// WMMA GEMM: C[M,Nc] = act(X[M,K=128] @ W[128,Nc] + bias)
// Block = 8 waves. Block tile = 128 rows x 64 cols.
// Each wave: one 16-row tile x four 16-col tiles (register blocked),
// so each A fragment feeds 4 v_wmma ops. W slab staged once in LDS,
// transposed k-contiguous, so each B fragment is one 32B LDS read.
// act: 0 = none, 1 = relu
// ---------------------------------------------------------------------------
__global__ __launch_bounds__(256)
void gemm_wmma_f16(const float* __restrict__ X, const float* __restrict__ W,
                   float* __restrict__ C, const float* __restrict__ bias,
                   int M, int Nc, int act)
{
  __shared__ _Float16 wldsT[64 * LDS_STRIDE];      // 18 KB

  const int lane = threadIdx.x & 31;
  const int wave = threadIdx.x >> 5;
  const int colBase = blockIdx.y * 64;

  // ---- stage W[:, colBase:colBase+64] into LDS, transposed (k contiguous) ----
  for (int idx = threadIdx.x; idx < FD * 64; idx += 256) {
    int k = idx >> 6;                               // 0..127
    int c = idx & 63;                               // 0..63  (coalesced global)
    wldsT[c * LDS_STRIDE + k] = (_Float16)W[(size_t)k * Nc + colBase + c];
  }
  __syncthreads();

  const int nRowTiles = M >> 4;
  int rowTile = blockIdx.x * 8 + wave;
  if (rowTile >= nRowTiles) rowTile = nRowTiles - 1;   // clamp: barrier-safe,
  const int rowBase = rowTile << 4;                    // dup waves write same data
  const int hi   = lane >> 4;                      // lane half (0/1)
  const int lsub = lane & 15;

  v8f acc0 = {}, acc1 = {}, acc2 = {}, acc3 = {};
  const float* xrow = X + (size_t)(rowBase + lsub) * FD;

#pragma unroll
  for (int kk = 0; kk < FD; kk += 32) {
    // A fragment (16x32): lane holds row M=lsub; VGPR v, half lo ->
    //   k = kk + 16*hi + 2*v + lo   (matches B k order)
    v16h a;
#pragma unroll
    for (int v = 0; v < 8; ++v) {
      int k0 = kk + (hi << 4) + (v << 1);
      a[2 * v]     = (_Float16)xrow[k0];
      a[2 * v + 1] = (_Float16)xrow[k0 + 1];
    }
    // B fragments: halves j -> k = kk + 16*hi + j, column = colTile*16 + lsub;
    // k-contiguous in LDS -> single 32B read per fragment.
    const v16h b0 = *(const v16h*)&wldsT[(0 * 16 + lsub) * LDS_STRIDE + kk + (hi << 4)];
    const v16h b1 = *(const v16h*)&wldsT[(1 * 16 + lsub) * LDS_STRIDE + kk + (hi << 4)];
    const v16h b2 = *(const v16h*)&wldsT[(2 * 16 + lsub) * LDS_STRIDE + kk + (hi << 4)];
    const v16h b3 = *(const v16h*)&wldsT[(3 * 16 + lsub) * LDS_STRIDE + kk + (hi << 4)];
    acc0 = __builtin_amdgcn_wmma_f32_16x16x32_f16(false, a, false, b0, (short)0, acc0, false, false);
    acc1 = __builtin_amdgcn_wmma_f32_16x16x32_f16(false, a, false, b1, (short)0, acc1, false, false);
    acc2 = __builtin_amdgcn_wmma_f32_16x16x32_f16(false, a, false, b2, (short)0, acc2, false, false);
    acc3 = __builtin_amdgcn_wmma_f32_16x16x32_f16(false, a, false, b3, (short)0, acc3, false, false);
  }

  // C/D layout: VGPR r -> M = r + 8*hi, N = lsub (per 16-col tile)
#pragma unroll
  for (int ct = 0; ct < 4; ++ct) {
    const v8f& acc = ct == 0 ? acc0 : ct == 1 ? acc1 : ct == 2 ? acc2 : acc3;
    int col = colBase + ct * 16 + lsub;
    float bv = bias ? bias[col] : 0.0f;
#pragma unroll
    for (int r = 0; r < 8; ++r) {
      int m = rowBase + r + (hi << 3);
      float v = acc[r] + bv;
      if (act == 1) v = v > 0.0f ? v : 0.0f;
      C[(size_t)m * Nc + col] = v;
    }
  }
}

// ---------------------------------------------------------------------------
// Per-node attention logits: als[n,h] = <h[n,h,:], a_src[h,:]>, same for ald
// ---------------------------------------------------------------------------
__global__ void att_logits(const float* __restrict__ hfeat,
                           const float* __restrict__ a_src,
                           const float* __restrict__ a_dst,
                           float* __restrict__ als, float* __restrict__ ald)
{
  int t = blockIdx.x * blockDim.x + threadIdx.x;   // n*NHEAD + head
  if (t >= NNODES * NHEAD) return;
  int n = t >> 2, hd = t & 3;
  const float* row = hfeat + (size_t)n * FD + hd * DHEAD;
  const float* as  = a_src + hd * DHEAD;
  const float* ad  = a_dst + hd * DHEAD;
  float s = 0.f, d = 0.f;
#pragma unroll 8
  for (int i = 0; i < DHEAD; ++i) { s += row[i] * as[i]; d += row[i] * ad[i]; }
  als[t] = s; ald[t] = d;
}

// ---------------------------------------------------------------------------
// Init accumulators: xacc = 0, m = ord(-inf) (=0), denom = 0
// ---------------------------------------------------------------------------
__global__ void init_acc(float* __restrict__ xacc, unsigned* __restrict__ mmax,
                         float* __restrict__ denom)
{
  int t = blockIdx.x * blockDim.x + threadIdx.x;
  if (t < NNODES * FD) xacc[t] = 0.0f;
  if (t < NNODES * NHEAD) { mmax[t] = 0u; denom[t] = 0.0f; }
}

__device__ __forceinline__ void edge_sd(const int* __restrict__ ei, int i,
                                        int& s, int& d)
{
  if (i < NEDGES) { s = ei[i]; d = ei[NEDGES + i]; }
  else            { s = d = i - NEDGES; }          // self loops appended
}

// pass 1: segment max of leaky-relu logits
__global__ void edge_max(const int* __restrict__ ei,
                         const float* __restrict__ als,
                         const float* __restrict__ ald,
                         unsigned* __restrict__ mmax)
{
  int t = blockIdx.x * blockDim.x + threadIdx.x;   // i*NHEAD + head
  if (t >= (NEDGES + NNODES) * NHEAD) return;
  int i = t >> 2, hd = t & 3, s, d;
  edge_sd(ei, i, s, d);
  float e = als[s * NHEAD + hd] + ald[d * NHEAD + hd];
  e = e > 0.f ? e : SLOPE_ATT * e;
  atomicMax(&mmax[d * NHEAD + hd], f2ord(e));
}

// pass 2: segment sum of exp(e - m)
__global__ void edge_denom(const int* __restrict__ ei,
                           const float* __restrict__ als,
                           const float* __restrict__ ald,
                           const unsigned* __restrict__ mmax,
                           float* __restrict__ denom)
{
  int t = blockIdx.x * blockDim.x + threadIdx.x;
  if (t >= (NEDGES + NNODES) * NHEAD) return;
  int i = t >> 2, hd = t & 3, s, d;
  edge_sd(ei, i, s, d);
  float e = als[s * NHEAD + hd] + ald[d * NHEAD + hd];
  e = e > 0.f ? e : SLOPE_ATT * e;
  atomicAdd(&denom[d * NHEAD + hd], expf(e - ord2f(mmax[d * NHEAD + hd])));
}

// pass 3: weighted scatter-add, one wave per edge, 4 channels per lane
__global__ __launch_bounds__(256)
void edge_scatter(const int* __restrict__ ei,
                  const float* __restrict__ als,
                  const float* __restrict__ ald,
                  const unsigned* __restrict__ mmax,
                  const float* __restrict__ denom,
                  const float* __restrict__ hfeat,
                  float* __restrict__ xacc)
{
  int gw   = (int)((blockIdx.x * blockDim.x + threadIdx.x) >> 5); // edge id
  int lane = threadIdx.x & 31;
  if (gw >= NEDGES + NNODES) return;
  int s, d;
  edge_sd(ei, gw, s, d);
  int hd = lane >> 3;                               // head for channels 4*lane..
  float e = als[s * NHEAD + hd] + ald[d * NHEAD + hd];
  e = e > 0.f ? e : SLOPE_ATT * e;
  float alpha = expf(e - ord2f(mmax[d * NHEAD + hd])) / denom[d * NHEAD + hd];
  int c = lane << 2;                                // channel base (0..124)
  const float4 hv = *(const float4*)(hfeat + (size_t)s * FD + c);
  float* outp = xacc + (size_t)d * FD + c;
  atomicAdd(outp + 0, alpha * hv.x);
  atomicAdd(outp + 1, alpha * hv.y);
  atomicAdd(outp + 2, alpha * hv.z);
  atomicAdd(outp + 3, alpha * hv.w);
}

// bias + leaky(0.01), in place
__global__ void bias_leaky(float* __restrict__ x, const float* __restrict__ b)
{
  int t = blockIdx.x * blockDim.x + threadIdx.x;
  if (t >= NNODES * FD) return;
  float v = x[t] + b[t & (FD - 1)];
  x[t] = v > 0.f ? v : SLOPE_ACT * v;
}

// ---------------------------------------------------------------------------
extern "C" void kernel_launch(void* const* d_in, const int* in_sizes, int n_in,
                              void* d_out, int out_size, void* d_ws, size_t ws_size,
                              hipStream_t stream)
{
  const float* src   = (const float*)d_in[0];
  const int*   ei    = (const int*)  d_in[1];
  const float* W1    = (const float*)d_in[2];
  const float* asrc1 = (const float*)d_in[3];
  const float* adst1 = (const float*)d_in[4];
  const float* b1    = (const float*)d_in[5];
  const float* W2    = (const float*)d_in[6];
  const float* asrc2 = (const float*)d_in[7];
  const float* adst2 = (const float*)d_in[8];
  const float* b2    = (const float*)d_in[9];
  const float* Wl    = (const float*)d_in[10];
  const float* bl    = (const float*)d_in[11];
  float* out = (float*)d_out;

  // workspace layout (floats): h(25.6MB) | x(25.6MB) | m,denom,als,ald (3.2MB)
  float*    ws    = (float*)d_ws;
  float*    hbuf  = ws;                                   // [N,128]
  float*    xbuf  = ws + (size_t)NNODES * FD;             // [N,128]
  float*    small = ws + (size_t)2 * NNODES * FD;
  unsigned* mmax  = (unsigned*)small;                     // [N,4]
  float*    denom = small + (size_t)NNODES * NHEAD;       // [N,4]
  float*    als   = small + (size_t)2 * NNODES * NHEAD;   // [N,4]
  float*    ald   = small + (size_t)3 * NNODES * NHEAD;   // [N,4]

  const int rowTiles  = NNODES / 16;                      // 3125
  const int gemmGridX = (rowTiles + 7) / 8;               // 8 row tiles / block
  dim3 gemmBlk(256);
  dim3 gemmGrid128(gemmGridX, FD / 64);                   // 2 col blocks
  dim3 gemmGrid512(gemmGridX, DFFOUT / 64);               // 8 col blocks

  const int nh        = NNODES * NHEAD;
  const int nodeElems = NNODES * FD;
  const int edgeElems = (NEDGES + NNODES) * NHEAD;
  const int nScatterBlocks = ((NEDGES + NNODES) + 7) / 8; // 8 waves/block
  dim3 blk256(256);
  dim3 gLogit((nh + 255) / 256);
  dim3 gInit((nodeElems + 255) / 256);
  dim3 gEdge((edgeElems + 255) / 256);
  dim3 gScat(nScatterBlocks);

  // ---------------- layer 1 ----------------
  gemm_wmma_f16<<<gemmGrid128, gemmBlk, 0, stream>>>(src, W1, hbuf, nullptr,
                                                     NNODES, FD, 0);
  att_logits<<<gLogit, blk256, 0, stream>>>(hbuf, asrc1, adst1, als, ald);
  init_acc<<<gInit, blk256, 0, stream>>>(xbuf, mmax, denom);
  edge_max<<<gEdge, blk256, 0, stream>>>(ei, als, ald, mmax);
  edge_denom<<<gEdge, blk256, 0, stream>>>(ei, als, ald, mmax, denom);
  edge_scatter<<<gScat, blk256, 0, stream>>>(ei, als, ald, mmax, denom, hbuf, xbuf);
  bias_leaky<<<gInit, blk256, 0, stream>>>(xbuf, b1);

  // ---------------- layer 2 ----------------
  gemm_wmma_f16<<<gemmGrid128, gemmBlk, 0, stream>>>(xbuf, W2, hbuf, nullptr,
                                                     NNODES, FD, 0);
  att_logits<<<gLogit, blk256, 0, stream>>>(hbuf, asrc2, adst2, als, ald);
  init_acc<<<gInit, blk256, 0, stream>>>(xbuf, mmax, denom);
  edge_max<<<gEdge, blk256, 0, stream>>>(ei, als, ald, mmax);
  edge_denom<<<gEdge, blk256, 0, stream>>>(ei, als, ald, mmax, denom);
  edge_scatter<<<gScat, blk256, 0, stream>>>(ei, als, ald, mmax, denom, hbuf, xbuf);
  bias_leaky<<<gInit, blk256, 0, stream>>>(xbuf, b2);

  // ---------------- final FFN: relu(x @ Wl + bl) ----------------
  gemm_wmma_f16<<<gemmGrid512, gemmBlk, 0, stream>>>(xbuf, Wl, out, bl,
                                                     NNODES, DFFOUT, 1);
}
